// mini_eMLP_74208444940656
// MI455X (gfx1250) — matrix operand order repaired and checked by reading, not compiled, not run on hardware
//
#include <hip/hip_runtime.h>

typedef __attribute__((ext_vector_type(16))) _Float16 v16h;
typedef __attribute__((ext_vector_type(8)))  float    v8f;

#define BATCH 256
#define TSTEPS 100
#define DIN  2312
#define HID  512
#define OUTN 11
#define KT_IN_FULL 72  // full 32-wide K tiles of DIN
#define KT_H  16       // 512/32
#define NT_H  32       // 512/16
#define MT_IN 1600     // 25600/16

__device__ __forceinline__ v8f wmma_f16(v16h a, v16h b, v8f c) {
    return __builtin_amdgcn_wmma_f32_16x16x32_f16(
        /*neg_a=*/false, a, /*neg_b=*/false, b,
        /*c_mod=*/(short)0, c, /*reuse_a=*/false, /*reuse_b=*/false);
}

// Unguarded 16-bit A-fragment build from an f32 row (16x32 tile).
// Lanes 0-15: K = kb..kb+7 and kb+16..kb+23 with kb = k0; lanes 16-31: kb = k0+8.
__device__ __forceinline__ v16h load_a16(const float* __restrict__ row, int kb) {
    v16h a;
#pragma unroll
    for (int j = 0; j < 8; ++j) {
        a[j]     = (_Float16)row[kb + j];
        a[8 + j] = (_Float16)row[kb + 16 + j];
    }
    return a;
}

// ---- Pack W[N,K] (row-major) into WMMA-ready B tiles -----------------------
// Tile (kt, nt) holds B[k][n] = W[n][k]; per lane 16 f16:
// lanes 0-15 -> K = k0..k0+7,k0+16..k0+23, lanes 16-31 -> K = k0+8..k0+15,k0+24..k0+31, N = n0+(lane&15)
__global__ void pack_b(const float* __restrict__ W, _Float16* __restrict__ out,
                       int N, int K, int Ktiles, int Ntiles) {
    int gid = blockIdx.x * blockDim.x + threadIdx.x;
    int total = Ktiles * Ntiles * 32;
    if (gid >= total) return;
    int lane = gid & 31;
    int tile = gid >> 5;
    int nt = tile % Ntiles;
    int kt = tile / Ntiles;
    int n = nt * 16 + (lane & 15);
    int kbase = kt * 32 + ((lane & 16) ? 8 : 0);
    _Float16* dst = out + (size_t)tile * 512 + lane * 16;
#pragma unroll
    for (int j = 0; j < 16; ++j) {
        int k = kbase + (j < 8 ? j : j + 8);
        float v = (n < N && k < K) ? W[(size_t)n * K + k] : 0.0f;
        dst[j] = (_Float16)v;
    }
}

__global__ void zero_f32(float* __restrict__ p, int n) {
    int i = blockIdx.x * blockDim.x + threadIdx.x;
    if (i < n) p[i] = 0.0f;
}

// ---- Big time-independent GEMM: I[m][h] = x[m]·Wi[h] + bi[h] + bh[h] -------
// m = b*T + t over 25600 rows. Block = 8 waves; wave w covers 64 cols.
__global__ __launch_bounds__(256) void gemm_input(
    const float* __restrict__ x,        // [25600, 2312]
    const _Float16* __restrict__ WiB,   // packed B tiles [73][32]
    const float* __restrict__ bi, const float* __restrict__ bh,
    float* __restrict__ I) {            // [25600, 512]
    int wave = threadIdx.x >> 5;
    int lane = threadIdx.x & 31;
    int m0 = blockIdx.x * 16;
    int row = m0 + (lane & 15);
    const float* arow = x + (size_t)row * DIN;
    int aoff = (lane & 16) ? 8 : 0;
    int ntb = wave * 4;                 // first n-tile of this wave

    v8f acc[4] = {};
    for (int kt = 0; kt < KT_IN_FULL; ++kt) {
        v16h a = load_a16(arow, kt * 32 + aoff);
        const _Float16* bt = WiB + ((size_t)kt * NT_H + ntb) * 512 + lane * 16;
#pragma unroll
        for (int nt = 0; nt < 4; ++nt) {
            v16h b = *(const v16h*)(bt + nt * 512);
            acc[nt] = wmma_f16(a, b, acc[nt]);
        }
    }
    // K tail: cols 2304..2311 (DIN % 32 == 8). Only lanes 0-15, first 8-run
    // are valid; packed B already zero-pads K >= DIN.
    {
        v16h a = {};
        if (!(lane & 16)) {
#pragma unroll
            for (int j = 0; j < 8; ++j)
                a[j] = (_Float16)arow[KT_IN_FULL * 32 + j];
        }
        const _Float16* bt =
            WiB + ((size_t)KT_IN_FULL * NT_H + ntb) * 512 + lane * 16;
#pragma unroll
        for (int nt = 0; nt < 4; ++nt) {
            v16h b = *(const v16h*)(bt + nt * 512);
            acc[nt] = wmma_f16(a, b, acc[nt]);
        }
    }
    int col = lane & 15;
    int r0 = (lane & 16) ? 8 : 0;
#pragma unroll
    for (int nt = 0; nt < 4; ++nt) {
        int n = (ntb + nt) * 16 + col;
        float bias = bi[n] + bh[n];
#pragma unroll
        for (int r = 0; r < 8; ++r) {
            int m = m0 + r0 + r;
            I[(size_t)m * HID + n] = acc[nt][r] + bias;
        }
    }
}

// ---- Recurrent step: h_in = I[t] + spk_in·Wh^T; membrane update ------------
__global__ __launch_bounds__(256) void recur_step(
    int t,
    const float* __restrict__ I,        // [25600, 512]
    const float* __restrict__ spk_in,   // [256, 512]
    float* __restrict__ spk_out,        // [256, 512]
    float* __restrict__ mem_h,          // [256, 512]
    const _Float16* __restrict__ WhB,   // packed [16][32]
    const float* __restrict__ thr_h,
    float* __restrict__ events) {
    int wave = threadIdx.x >> 5;
    int lane = threadIdx.x & 31;
    int m0 = blockIdx.x * 16;
    int row = m0 + (lane & 15);
    const float* arow = spk_in + (size_t)row * HID;
    int aoff = (lane & 16) ? 8 : 0;
    int ntb = wave * 4;

    v8f acc[4] = {};
    for (int kt = 0; kt < KT_H; ++kt) {
        v16h a = load_a16(arow, kt * 32 + aoff);
        const _Float16* bt = WhB + ((size_t)kt * NT_H + ntb) * 512 + lane * 16;
#pragma unroll
        for (int nt = 0; nt < 4; ++nt) {
            v16h b = *(const v16h*)(bt + nt * 512);
            acc[nt] = wmma_f16(a, b, acc[nt]);
        }
    }
    int col = lane & 15;
    int r0 = (lane & 16) ? 8 : 0;
    float spksum = 0.0f;
#pragma unroll
    for (int nt = 0; nt < 4; ++nt) {
        int n = (ntb + nt) * 16 + col;
        float thr = thr_h[n];
#pragma unroll
        for (int r = 0; r < 8; ++r) {
            int b = m0 + r0 + r;
            float hin = acc[nt][r] + I[((size_t)b * TSTEPS + t) * HID + n];
            float mem = mem_h[(size_t)b * HID + n] + hin;   // DECAY = 1
            float spk = (mem > thr) ? 1.0f : 0.0f;
            mem = (spk != 0.0f) ? 0.0f : mem;
            mem = (mem < -thr) ? -thr : mem;
            mem_h[(size_t)b * HID + n] = mem;
            spk_out[(size_t)b * HID + n] = spk;
            spksum += spk;
        }
    }
    atomicAdd(&events[t], spksum);   // integer-valued floats: order-exact
}

// ---- Output layer: o_in = spk·Wo^T + bo; membrane + spike_sum --------------
__global__ __launch_bounds__(512) void out_step(
    int t,
    const float* __restrict__ spk,      // [256, 512] (spikes at t)
    const _Float16* __restrict__ WoB,   // packed [16][1]
    const float* __restrict__ bo,
    const float* __restrict__ thr_o,
    float* __restrict__ mem_o,          // [256, 16]
    float* __restrict__ spike_sum,      // [256, 16]
    float* __restrict__ events) {
    int wave = threadIdx.x >> 5;        // 16 waves, one 16-row tile each
    int lane = threadIdx.x & 31;
    int m0 = wave * 16;
    int row = m0 + (lane & 15);
    const float* arow = spk + (size_t)row * HID;
    int aoff = (lane & 16) ? 8 : 0;

    v8f acc = {};
    for (int kt = 0; kt < KT_H; ++kt) {
        v16h a = load_a16(arow, kt * 32 + aoff);
        v16h b = *(const v16h*)(WoB + (size_t)kt * 512 + lane * 16);
        acc = wmma_f16(a, b, acc);
    }
    int n = lane & 15;
    int r0 = (lane & 16) ? 8 : 0;
    if (n < OUTN) {
        float thr = thr_o[n];
        float bb = bo[n];
        float s = 0.0f;
#pragma unroll
        for (int r = 0; r < 8; ++r) {
            int b = m0 + r0 + r;
            float mem = mem_o[b * 16 + n] + acc[r] + bb;
            float spkv = (mem > thr) ? 1.0f : 0.0f;
            mem = (spkv != 0.0f) ? 0.0f : mem;
            mem = (mem < -thr) ? -thr : mem;
            mem_o[b * 16 + n] = mem;
            spike_sum[b * 16 + n] += spkv;
            s += spkv;
        }
        atomicAdd(&events[t], s);
    }
}

// ---- Final: log-softmax, argmax, NLL loss, event averages ------------------
__global__ __launch_bounds__(256) void finalize_k(
    const float* __restrict__ spike_sum, const int* __restrict__ labels,
    const float* __restrict__ events, float* __restrict__ out) {
    __shared__ float red[256];
    int b = threadIdx.x;
    const float* v = spike_sum + b * 16;
    float m = v[0]; int arg = 0;
    for (int j = 1; j < OUTN; ++j) { if (v[j] > m) { m = v[j]; arg = j; } }
    float se = 0.0f;
    for (int j = 0; j < OUTN; ++j) se += __expf(v[j] - m);
    float lse = __logf(se) + m;
    out[b] = (float)arg;                // predictions [1,256]
    red[b] = lse - v[labels[b]];        // -log p[label]
    __syncthreads();
    for (int s = 128; s > 0; s >>= 1) {
        if (b < s) red[b] += red[b + s];
        __syncthreads();
    }
    if (b == 0) out[BATCH] = red[0] * (1.0f / BATCH);          // loss
    if (b < TSTEPS) out[BATCH + 1 + b] = events[b] * (1.0f / BATCH);
}

extern "C" void kernel_launch(void* const* d_in, const int* in_sizes, int n_in,
                              void* d_out, int out_size, void* d_ws, size_t ws_size,
                              hipStream_t stream) {
    const float* x      = (const float*)d_in[0];   // [256,100,2312]
    const int*   labels = (const int*)  d_in[1];   // [256]
    // d_in[2..5]: initial states (zeros) — we re-zero our own copies
    const float* Wi     = (const float*)d_in[6];   // [512,2312]
    const float* bi     = (const float*)d_in[7];
    const float* Wh     = (const float*)d_in[8];   // [512,512]
    const float* bh     = (const float*)d_in[9];
    const float* Wo     = (const float*)d_in[10];  // [11,512]
    const float* bo     = (const float*)d_in[11];
    const float* thr_h  = (const float*)d_in[12];
    const float* thr_o  = (const float*)d_in[13];
    float* out = (float*)d_out;

    char* base = (char*)d_ws;
    size_t off = 0;
    auto carve = [&](size_t bytes) {
        char* p = base + off;
        off = (off + bytes + 255) & ~(size_t)255;
        return p;
    };
    const int KT_IN = KT_IN_FULL + 1;  // 73 packed K tiles (last zero-padded)
    float*    I    = (float*)   carve((size_t)25600 * HID * 4);       // 52.4 MB
    _Float16* WiB  = (_Float16*)carve((size_t)KT_IN * NT_H * 1024);   // 2.4 MB
    _Float16* WhB  = (_Float16*)carve((size_t)KT_H  * NT_H * 1024);   // 512 KB
    _Float16* WoB  = (_Float16*)carve((size_t)KT_H  * 1    * 1024);   // 16 KB
    // contiguous zeroed state region:
    float* state0    = (float*)(base + off);
    float* mem_h     = (float*)carve((size_t)BATCH * HID * 4);
    float* spk0      = (float*)carve((size_t)BATCH * HID * 4);
    float* spk1      = (float*)carve((size_t)BATCH * HID * 4);
    float* mem_o     = (float*)carve((size_t)BATCH * 16 * 4);
    float* spike_sum = (float*)carve((size_t)BATCH * 16 * 4);
    float* events    = (float*)carve(512);
    int stateFloats = (int)(((base + off) - (char*)state0) / 4);

    // 1) pack weights into WMMA B-tile layout
    {
        int tot = KT_IN * NT_H * 32;
        pack_b<<<(tot + 255) / 256, 256, 0, stream>>>(Wi, WiB, HID, DIN, KT_IN, NT_H);
        tot = KT_H * NT_H * 32;
        pack_b<<<(tot + 255) / 256, 256, 0, stream>>>(Wh, WhB, HID, HID, KT_H, NT_H);
        tot = KT_H * 1 * 32;
        pack_b<<<(tot + 255) / 256, 256, 0, stream>>>(Wo, WoB, OUTN, HID, KT_H, 1);
    }
    // 2) zero all state (d_ws is poisoned; must be deterministic per call)
    zero_f32<<<(stateFloats + 255) / 256, 256, 0, stream>>>(state0, stateFloats);

    // 3) hoisted time-independent GEMM: I = x·Wi^T + (bi + bh)
    gemm_input<<<MT_IN, 256, 0, stream>>>(x, WiB, bi, bh, I);

    // 4) serial recurrence, double-buffered spikes
    for (int t = 0; t < TSTEPS; ++t) {
        const float* sin = (t & 1) ? spk1 : spk0;
        float*       sot = (t & 1) ? spk0 : spk1;
        recur_step<<<16, 256, 0, stream>>>(t, I, sin, sot, mem_h, WhB, thr_h, events);
        out_step<<<1, 512, 0, stream>>>(t, sot, WoB, bo, thr_o, mem_o, spike_sum, events);
    }

    // 5) predictions + loss + event averages
    finalize_k<<<1, 256, 0, stream>>>(spike_sum, labels, events, out);
    (void)in_sizes; (void)n_in; (void)out_size; (void)ws_size;
}
